// GAT_86045374808682
// MI455X (gfx1250) — matrix operand order repaired and checked by reading, not compiled, not run on hardware
//
#include <hip/hip_runtime.h>
#include <hip/hip_bf16.h>

typedef __attribute__((ext_vector_type(16))) _Float16 v16h;
typedef __attribute__((ext_vector_type(8)))  _Float16 v8h;
typedef __attribute__((ext_vector_type(2)))  _Float16 v2h;
typedef __attribute__((ext_vector_type(8)))  float    v8f;

#define NN   8192
#define INF_ 8193
#define HC_  128
#define HH   8
#define EE   262144
#define ETOT (EE + NN)

struct HPair { v8h lo; v8h hi; };
__device__ inline v16h combine16(v8h lo, v8h hi) {
    HPair p{lo, hi};
    return __builtin_bit_cast(v16h, p);
}
__device__ inline unsigned pack2(float a, float b) {
    v2h p = { (_Float16)a, (_Float16)b };
    return __builtin_bit_cast(unsigned, p);
}

// ---------------------------------------------------------------------------
// WMMA GEMM: C[M x N] = A[M x K(lda)] * B[K x N], f32 in/out, f16 WMMA math.
// NT = N/16 (8 -> N=128, 4 -> N=64). Split-K via (blockIdx.y, kchunk) with
// ATOMIC accumulate. 256 thr = 8 waves; wave owns a 16 x N strip of a 128-row
// tile. Double-buffered LDS, f16 fragment-order layout:
//   As[row][k]  (row stride 40 halfs)  -> A frag = 2x ds_load_b128
//   Bs[col][k]  (col stride 40 halfs)  -> B frag = 2x ds_load_b128
// ---------------------------------------------------------------------------
template <int NT, bool ATOMIC>
__global__ __launch_bounds__(256) void gemm_wmma(
    const float* __restrict__ A, const float* __restrict__ B,
    const float* __restrict__ bias, float* __restrict__ C,
    int M, int K, int lda, int kchunk)
{
    constexpr int N    = NT * 16;
    constexpr int AP   = 40;          // halfs per A row (32 + pad), 80B: 16B-aligned
    constexpr int BPk  = 40;          // halfs per B col
    constexpr int NGRP = 256 / N;     // B loader groups (2 or 4)
    constexpr int KPT  = 32 / NGRP;   // k per B-loader thread (16 or 8)
    __shared__ __align__(16) _Float16 As[2 * 128 * AP];
    __shared__ __align__(16) _Float16 Bs[2 * N * BPk];

    const int t    = threadIdx.x;
    const int wave = t >> 5;
    const int lane = t & 31;
    const int half = lane >> 4;
    const int l16  = lane & 15;
    const int m0   = blockIdx.x * 128;
    const int mw   = m0 + wave * 16;

    const int k0 = blockIdx.y * kchunk;
    int k1 = k0 + kchunk; if (k1 > K) k1 = K;
    const int kAligned = k0 + ((k1 - k0) & ~31);

    // loader coordinates
    const int rowA = t & 127;
    const int cgA  = t >> 7;          // 0..1, 16 k each
    const int nB   = t % N;
    const int gB   = t / N;           // k group

    float aReg[16];
    float bReg[KPT];

    auto gload = [&](int kk) {
        const float* ap = A + (size_t)(m0 + rowA) * lda + kk + cgA * 16;
        #pragma unroll
        for (int i = 0; i < 16; ++i) aReg[i] = ap[i];
        const float* bp = B + (size_t)(kk + gB * KPT) * N + nB;
        #pragma unroll
        for (int i = 0; i < KPT; ++i) bReg[i] = bp[(size_t)i * N];
    };
    auto lstore = [&](int b) {
        uint4* ad = (uint4*)&As[b * 128 * AP + rowA * AP + cgA * 16];
        #pragma unroll
        for (int q = 0; q < 2; ++q) {
            uint4 w;
            w.x = pack2(aReg[q*8+0], aReg[q*8+1]);
            w.y = pack2(aReg[q*8+2], aReg[q*8+3]);
            w.z = pack2(aReg[q*8+4], aReg[q*8+5]);
            w.w = pack2(aReg[q*8+6], aReg[q*8+7]);
            ad[q] = w;
        }
        uint4* bd = (uint4*)&Bs[b * N * BPk + nB * BPk + gB * KPT];
        #pragma unroll
        for (int q = 0; q < KPT / 8; ++q) {
            uint4 w;
            w.x = pack2(bReg[q*8+0], bReg[q*8+1]);
            w.y = pack2(bReg[q*8+2], bReg[q*8+3]);
            w.z = pack2(bReg[q*8+4], bReg[q*8+5]);
            w.w = pack2(bReg[q*8+6], bReg[q*8+7]);
            bd[q] = w;
        }
    };

    v8f acc[NT] = {};

    auto compute = [&](int b) {
        const _Float16* Ab = &As[b * 128 * AP];
        const _Float16* Bb = &Bs[b * N * BPk];
        const _Float16* arow = &Ab[(wave * 16 + l16) * AP + half * 8];
        v16h af = combine16(*(const v8h*)arow, *(const v8h*)(arow + 16));
        #pragma unroll
        for (int nt = 0; nt < NT; ++nt) {
            const _Float16* bcol = &Bb[(nt * 16 + l16) * BPk + half * 8];
            v16h bf = combine16(*(const v8h*)bcol, *(const v8h*)(bcol + 16));
            acc[nt] = __builtin_amdgcn_wmma_f32_16x16x32_f16(
                false, af, false, bf, (short)0, acc[nt], false, false);
        }
    };

    const int nsteps = (kAligned - k0) >> 5;
    if (nsteps > 0) { gload(k0); lstore(0); }
    for (int s = 0; s < nsteps; ++s) {
        __syncthreads();
        if (s + 1 < nsteps) gload(k0 + (s + 1) * 32);   // prefetch next tile
        compute(s & 1);                                 // WMMA while loads fly
        if (s + 1 < nsteps) lstore((s + 1) & 1);
    }

    // scalar K remainder (handles K=8193)
    for (int k = kAligned; k < k1; ++k) {
        float av[8];
        #pragma unroll
        for (int i = 0; i < 8; ++i)
            av[i] = A[(size_t)(mw + i + half * 8) * lda + k];
        #pragma unroll
        for (int nt = 0; nt < NT; ++nt) {
            float bv = B[(size_t)k * N + nt * 16 + l16];
            #pragma unroll
            for (int i = 0; i < 8; ++i) acc[nt][i] += av[i] * bv;
        }
    }

    #pragma unroll
    for (int nt = 0; nt < NT; ++nt) {
        int col = nt * 16 + l16;
        float bv = (!ATOMIC && bias) ? bias[col] : 0.f;
        #pragma unroll
        for (int i = 0; i < 8; ++i) {
            int row = mw + i + half * 8;
            float* cp = C + (size_t)row * N + col;
            if (ATOMIC) atomicAdd(cp, acc[nt][i]);
            else        *cp = acc[nt][i] + bv;
        }
    }
}

// ---------------------------------------------------------------------------
__global__ void fill_kernel(float* p, float v, int n) {
    int i = blockIdx.x * blockDim.x + threadIdx.x;
    if (i < n) p[i] = v;
}

__global__ void scores_kernel(const float* __restrict__ h,
                              const float* __restrict__ asrc,
                              const float* __restrict__ adst,
                              float* __restrict__ sa, float* __restrict__ da, int n)
{
    int i = blockIdx.x * blockDim.x + threadIdx.x;
    if (i >= n) return;
    int node = i >> 3, hd = i & 7;
    const float* hp = h + (size_t)node * HC_ + hd * 16;
    const float* ap = asrc + hd * 16;
    const float* bp = adst + hd * 16;
    float s = 0.f, d = 0.f;
    #pragma unroll
    for (int j = 0; j < 16; ++j) { float v = hp[j]; s += v * ap[j]; d += v * bp[j]; }
    sa[i] = s; da[i] = d;
}

__device__ inline void atomicMaxF(float* addr, float val) {
    if (val >= 0.f) atomicMax((int*)addr, __float_as_int(val));
    else            atomicMin((unsigned int*)addr, __float_as_uint(val));
}

__device__ inline void edge_sd(int e, const int* src, const int* dst, int& s, int& d) {
    if (e < EE) { s = src[e]; d = dst[e]; } else { s = d = e - EE; }
}

__global__ void edge_max_kernel(const int* __restrict__ src, const int* __restrict__ dst,
                                const float* __restrict__ sa, const float* __restrict__ da,
                                float* __restrict__ m)
{
    int i = blockIdx.x * blockDim.x + threadIdx.x;
    if (i >= ETOT * HH) return;
    int e = i >> 3, h = i & 7, s, d;
    edge_sd(e, src, dst, s, d);
    float v = sa[s * 8 + h] + da[d * 8 + h];
    v = v > 0.f ? v : v * 0.2f;
    atomicMaxF(&m[d * 8 + h], v);
}

__global__ void edge_sum_kernel(const int* __restrict__ src, const int* __restrict__ dst,
                                const float* __restrict__ sa, const float* __restrict__ da,
                                const float* __restrict__ m, float* __restrict__ exb,
                                float* __restrict__ den)
{
    int i = blockIdx.x * blockDim.x + threadIdx.x;
    if (i >= ETOT * HH) return;
    int e = i >> 3, h = i & 7, s, d;
    edge_sd(e, src, dst, s, d);
    float v = sa[s * 8 + h] + da[d * 8 + h];
    v = v > 0.f ? v : v * 0.2f;
    float ex = __expf(v - m[d * 8 + h]);
    exb[i] = ex;
    atomicAdd(&den[d * 8 + h], ex);
}

__global__ __launch_bounds__(256) void edge_agg_kernel(
    const int* __restrict__ src, const int* __restrict__ dst,
    const float* __restrict__ h, const float* __restrict__ exb,
    const float* __restrict__ den, float* __restrict__ out)
{
    long gid = (long)blockIdx.x * blockDim.x + threadIdx.x;
    long e = gid >> 5;
    int lane = (int)(gid & 31);
    if (e >= ETOT) return;
    int s, d;
    edge_sd((int)e, src, dst, s, d);
    int hd = lane >> 2;
    float alpha = exb[e * 8 + hd] / (den[d * 8 + hd] + 1e-16f);
    float4 v = ((const float4*)(h + (size_t)s * HC_))[lane];
    float* op = out + (size_t)d * HC_ + lane * 4;
    atomicAdd(op + 0, v.x * alpha);
    atomicAdd(op + 1, v.y * alpha);
    atomicAdd(op + 2, v.z * alpha);
    atomicAdd(op + 3, v.w * alpha);
}

template <int W>
__global__ __launch_bounds__(256) void post_ln_kernel(
    const float* __restrict__ in, const float* __restrict__ bias,
    const float* __restrict__ g, const float* __restrict__ be,
    const float* __restrict__ res, float* __restrict__ out, int n)
{
    int wid  = (blockIdx.x * blockDim.x + threadIdx.x) >> 5;
    int lane = threadIdx.x & 31;
    if (wid >= n) return;
    constexpr int V = W / 32;
    float v[V];
    const float* ip = in + (size_t)wid * W;
    float s = 0.f;
    #pragma unroll
    for (int i = 0; i < V; ++i) {
        int c = lane + i * 32;
        v[i] = ip[c] + (bias ? bias[c] : 0.f);
        s += v[i];
    }
    for (int off = 16; off; off >>= 1) s += __shfl_xor(s, off, 32);
    float mu = s / (float)W;
    float q = 0.f;
    #pragma unroll
    for (int i = 0; i < V; ++i) { float dd = v[i] - mu; q += dd * dd; }
    for (int off = 16; off; off >>= 1) q += __shfl_xor(q, off, 32);
    float inv = rsqrtf(q / (float)W + 1e-5f);
    float* op = out + (size_t)wid * W;
    #pragma unroll
    for (int i = 0; i < V; ++i) {
        int c = lane + i * 32;
        float y = (v[i] - mu) * inv * g[c] + be[c];
        y = y > 0.f ? y : 0.f;
        if (res) y += res[(size_t)wid * W + c];
        op[c] = y;
    }
}

__global__ __launch_bounds__(256) void rownorm_kernel(float* h, int n) {
    int wid  = (blockIdx.x * blockDim.x + threadIdx.x) >> 5;
    int lane = threadIdx.x & 31;
    if (wid >= n) return;
    float* p = h + (size_t)wid * HC_;
    float v[4], q = 0.f;
    #pragma unroll
    for (int i = 0; i < 4; ++i) { v[i] = p[lane + i * 32]; q += v[i] * v[i]; }
    for (int off = 16; off; off >>= 1) q += __shfl_xor(q, off, 32);
    float nrm = sqrtf(q); nrm = nrm > 1e-12f ? nrm : 1e-12f;
    float inv = 1.f / nrm;
    #pragma unroll
    for (int i = 0; i < 4; ++i) p[lane + i * 32] = v[i] * inv;
}

__global__ __launch_bounds__(256) void head_kernel(
    const float* __restrict__ a_hid, const float* __restrict__ aW2, const float* __restrict__ ab2,
    const float* __restrict__ r_hid, const float* __restrict__ rW2, const float* __restrict__ rb2,
    float* __restrict__ coords, float* __restrict__ sums, int n)
{
    int wid  = (blockIdx.x * blockDim.x + threadIdx.x) >> 5;
    int lane = threadIdx.x & 31;
    if (wid >= n) return;
    float s1 = 0.f, s2 = 0.f;
    #pragma unroll
    for (int i = 0; i < 4; ++i) { int c = lane + i * 32; s1 += a_hid[(size_t)wid * 128 + c] * aW2[c]; }
    #pragma unroll
    for (int i = 0; i < 2; ++i) { int c = lane + i * 32; s2 += r_hid[(size_t)wid * 64 + c] * rW2[c]; }
    for (int off = 16; off; off >>= 1) { s1 += __shfl_xor(s1, off, 32); s2 += __shfl_xor(s2, off, 32); }
    if (lane == 0) {
        float ang = tanhf(s1 + ab2[0]) * 3.14159265358979323846f;
        float x = s2 + rb2[0];
        float sp = x > 20.f ? x : log1pf(__expf(x));
        float r = 1.f + 0.1f * tanhf(sp);
        float cx = r * cosf(ang), cy = r * sinf(ang);
        coords[wid * 2] = cx; coords[wid * 2 + 1] = cy;
        atomicAdd(&sums[0], cx); atomicAdd(&sums[1], cy);
    }
}

__global__ void final_kernel(const float* __restrict__ coords, const float* __restrict__ sums,
                             float* __restrict__ out, int n)
{
    int i = blockIdx.x * blockDim.x + threadIdx.x;
    if (i >= n) return;
    float cx = coords[i * 2]     - sums[0] / (float)n;
    float cy = coords[i * 2 + 1] - sums[1] / (float)n;
    float nrm = sqrtf(cx * cx + cy * cy);
    nrm = nrm > 1e-12f ? nrm : 1e-12f;
    out[i * 2] = cx / nrm; out[i * 2 + 1] = cy / nrm;
}

// ---------------------------------------------------------------------------
extern "C" void kernel_launch(void* const* d_in, const int* in_sizes, int n_in,
                              void* d_out, int out_size, void* d_ws, size_t ws_size,
                              hipStream_t stream)
{
    (void)in_sizes; (void)n_in; (void)out_size; (void)ws_size;
    const float* x    = (const float*)d_in[0];
    const int*   srcp = (const int*)d_in[1];
    const int*   dstp = (const int*)d_in[2];
    const float* W1  = (const float*)d_in[3];
    const float* as1 = (const float*)d_in[4];
    const float* ad1 = (const float*)d_in[5];
    const float* b1  = (const float*)d_in[6];
    const float* W2  = (const float*)d_in[7];
    const float* as2 = (const float*)d_in[8];
    const float* ad2 = (const float*)d_in[9];
    const float* b2  = (const float*)d_in[10];
    const float* W3  = (const float*)d_in[11];
    const float* as3 = (const float*)d_in[12];
    const float* ad3 = (const float*)d_in[13];
    const float* b3  = (const float*)d_in[14];
    const float* g1  = (const float*)d_in[15];
    const float* be1 = (const float*)d_in[16];
    const float* g2  = (const float*)d_in[17];
    const float* be2 = (const float*)d_in[18];
    const float* g3  = (const float*)d_in[19];
    const float* be3 = (const float*)d_in[20];
    const float* aW1 = (const float*)d_in[21];
    const float* ab1 = (const float*)d_in[22];
    const float* ag  = (const float*)d_in[23];
    const float* abe = (const float*)d_in[24];
    const float* aW2 = (const float*)d_in[25];
    const float* ab2 = (const float*)d_in[26];
    const float* rW1 = (const float*)d_in[27];
    const float* rb1 = (const float*)d_in[28];
    const float* rg  = (const float*)d_in[29];
    const float* rbe = (const float*)d_in[30];
    const float* rW2 = (const float*)d_in[31];
    const float* rb2 = (const float*)d_in[32];
    float* out = (float*)d_out;

    float* ws   = (float*)d_ws;
    float* bufA = ws;                         // N*128 (projected h / a_hid)
    float* bufB = ws + 1048576;               // N*128 (aggregated / r_hid)
    float* h1   = ws + 2 * 1048576;
    float* h2   = ws + 3 * 1048576;
    float* h3   = ws + 4 * 1048576;
    float* sa   = ws + 5 * 1048576;           // N*8
    float* da   = sa + 65536;
    float* mbuf = da + 65536;
    float* den  = mbuf + 65536;
    float* exb  = den + 65536;                // ETOT*8 = 2162688
    float* crd  = exb + 2162688;              // N*2
    float* sums = crd + 16384;                // 2

    const int eth = ETOT * HH;
    const long aggThreads = (long)ETOT * 32;

    auto edge_phase = [&](const float* hfeat, const float* asr, const float* ads) {
        scores_kernel<<<(NN * HH + 255) / 256, 256, 0, stream>>>(hfeat, asr, ads, sa, da, NN * HH);
        fill_kernel<<<256, 256, 0, stream>>>(mbuf, -__builtin_inff(), 65536);
        fill_kernel<<<256, 256, 0, stream>>>(den, 0.f, 65536);
        fill_kernel<<<4096, 256, 0, stream>>>(bufB, 0.f, 1048576);
        edge_max_kernel<<<(eth + 255) / 256, 256, 0, stream>>>(srcp, dstp, sa, da, mbuf);
        edge_sum_kernel<<<(eth + 255) / 256, 256, 0, stream>>>(srcp, dstp, sa, da, mbuf, exb, den);
        edge_agg_kernel<<<(int)((aggThreads + 255) / 256), 256, 0, stream>>>(srcp, dstp, hfeat, exb, den, bufB);
    };

    // ---- layer 1: h = x @ W1 (K=8193, split-K=8, atomic accumulate) ----
    fill_kernel<<<4096, 256, 0, stream>>>(bufA, 0.f, 1048576);
    gemm_wmma<8, true><<<dim3(64, 8), 256, 0, stream>>>(x, W1, nullptr, bufA, NN, INF_, INF_, 1024);
    edge_phase(bufA, as1, ad1);
    post_ln_kernel<128><<<1024, 256, 0, stream>>>(bufB, b1, g1, be1, nullptr, h1, NN);

    // ---- layer 2 ----
    gemm_wmma<8, false><<<dim3(64, 1), 256, 0, stream>>>(h1, W2, nullptr, bufA, NN, HC_, HC_, HC_);
    edge_phase(bufA, as2, ad2);
    post_ln_kernel<128><<<1024, 256, 0, stream>>>(bufB, b2, g2, be2, h1, h2, NN);

    // ---- layer 3 ----
    gemm_wmma<8, false><<<dim3(64, 1), 256, 0, stream>>>(h2, W3, nullptr, bufA, NN, HC_, HC_, HC_);
    edge_phase(bufA, as3, ad3);
    post_ln_kernel<128><<<1024, 256, 0, stream>>>(bufB, b3, g3, be3, h2, h3, NN);

    // ---- row-normalize h3 ----
    rownorm_kernel<<<1024, 256, 0, stream>>>(h3, NN);

    // ---- angle head ----
    gemm_wmma<8, false><<<dim3(64, 1), 256, 0, stream>>>(h3, aW1, ab1, bufA, NN, HC_, HC_, HC_);
    post_ln_kernel<128><<<1024, 256, 0, stream>>>(bufA, nullptr, ag, abe, nullptr, bufA, NN);

    // ---- radius head (N=64) ----
    gemm_wmma<4, false><<<dim3(64, 1), 256, 0, stream>>>(h3, rW1, rb1, bufB, NN, HC_, HC_, HC_);
    post_ln_kernel<64><<<1024, 256, 0, stream>>>(bufB, nullptr, rg, rbe, nullptr, bufB, NN);

    // ---- coords + mean + final normalize ----
    fill_kernel<<<1, 256, 0, stream>>>(sums, 0.f, 2);
    head_kernel<<<1024, 256, 0, stream>>>(bufA, aW2, ab2, bufB, rW2, rb2, crd, sums, NN);
    final_kernel<<<(NN + 255) / 256, 256, 0, stream>>>(crd, sums, out, NN);
}
